// BrainFormerBlock_9887014716196
// MI455X (gfx1250) — matrix operand order repaired and checked by reading, not compiled, not run on hardware
//
#include <hip/hip_runtime.h>
#include <math.h>

typedef unsigned short u16;
typedef __bf16 bf16;
typedef bf16 v16bf __attribute__((ext_vector_type(16)));
typedef float v8f __attribute__((ext_vector_type(8)));
typedef unsigned int uint32x4 __attribute__((ext_vector_type(4)));
typedef int int32x4 __attribute__((ext_vector_type(4)));
typedef int int32x8 __attribute__((ext_vector_type(8)));

#if __has_builtin(__builtin_amdgcn_tensor_load_to_lds)
#define USE_TDM 1
#else
#define USE_TDM 0
#endif

// ---------------- bf16 helpers ----------------
__device__ inline u16 f2bf(float f) {
  unsigned u = __float_as_uint(f);
  unsigned r = u + 0x7FFFu + ((u >> 16) & 1u);   // round-to-nearest-even
  return (u16)(r >> 16);
}
__device__ inline float bf2f(u16 h) { return __uint_as_float(((unsigned)h) << 16); }

union FragAB { v16bf v; uint4 q[2]; };

// A-matrix fragment (16x32 bf16). Per ISA: lanes0-15 hold K0..7,K16..23; lanes16-31 K8..15,K24..31.
__device__ inline v16bf ld_fragA(const u16* base, int stride) {
  int lane = threadIdx.x & 31;
  const u16* p = base + (lane & 15) * stride + ((lane >> 4) << 3);
  FragAB f;
  f.q[0] = *(const uint4*)(p);
  f.q[1] = *(const uint4*)(p + 16);
  return f.v;
}
// B-matrix fragment (32x16 bf16), source stored transposed [n][k].
// Same K order as A: lanes0-15 hold K0..7,K16..23; lanes16-31 K8..15,K24..31.
__device__ inline v16bf ld_fragB(const u16* base, int stride) {
  int lane = threadIdx.x & 31;
  const u16* p = base + (lane & 15) * stride + ((lane >> 4) << 3);
  FragAB f;
  f.q[0] = *(const uint4*)(p);
  f.q[1] = *(const uint4*)(p + 16);
  return f.v;
}
__device__ inline v8f wmma_bf16(v16bf a, v16bf b, v8f c) {
  return __builtin_amdgcn_wmma_f32_16x16x32_bf16(false, a, false, b, (short)0, c, false, false);
}

#if USE_TDM
// TDM: DMA a [tile_d1 x tile_d0(contig)] bf16 tile (row stride = stride0 elems) into LDS,
// with HW padding 4 dwords after every 16 dwords -> LDS row stride 40 elems.
__device__ inline void tdm_load_2d(unsigned lds_off, const void* gaddr,
                                   unsigned tile_d0, unsigned tile_d1,
                                   unsigned long long stride0_elems) {
  unsigned long long ga = (unsigned long long)gaddr;
  uint32x4 g0;
  g0[0] = 1u;                                              // count=1, user descriptor
  g0[1] = lds_off;                                         // lds_addr (bytes)
  g0[2] = (unsigned)(ga & 0xFFFFFFFFu);                    // global_addr[31:0]
  g0[3] = (unsigned)((ga >> 32) & 0x01FFFFFFu) | (2u << 30); // global_addr[56:32], type=2
  int32x8 g1;
  g1[0] = (int)((1u << 16)      // data_size = 2 bytes
              | (1u << 20)      // pad_enable
              | (3u << 22)      // pad_interval: 16 dwords
              | (3u << 25));    // pad_amount: 4 dwords
  unsigned td0 = 0x7FFFFFFFu, td1 = 0x7FFFFFFFu;           // huge tensor dims (no OOB)
  g1[1] = (int)((td0 & 0xFFFFu) << 16);
  g1[2] = (int)(((td0 >> 16) & 0xFFFFu) | ((td1 & 0xFFFFu) << 16));
  g1[3] = (int)(((td1 >> 16) & 0xFFFFu) | ((tile_d0 & 0xFFFFu) << 16));
  g1[4] = (int)(tile_d1 & 0xFFFFu);                        // tile_dim1; tile_dim2=0
  g1[5] = (int)(stride0_elems & 0xFFFFFFFFu);
  g1[6] = (int)((stride0_elems >> 32) & 0xFFFFu);
  g1[7] = 0;
  int32x4 z4 = {0, 0, 0, 0};
#if defined(__clang_major__) && (__clang_major__ >= 23)
  int32x8 z8 = {0, 0, 0, 0, 0, 0, 0, 0};
  __builtin_amdgcn_tensor_load_to_lds(g0, g1, z4, z4, z8, 0);
#else
  __builtin_amdgcn_tensor_load_to_lds(g0, g1, z4, z4, 0);
#endif
}
#endif

// ---------------- GEMM: C[M,N] = A[M,K](bf16) x BT[N,K](bf16) (+res), batched over z ----------------
__global__ __launch_bounds__(256) void gemm_bf16_wmma(
    const u16* __restrict__ A, const u16* __restrict__ BT, float* __restrict__ C,
    int M, int N, int K, long sA, long sB, long sC, const float* __restrict__ res) {
  __shared__ u16 lA[2][128 * 40];
  __shared__ u16 lB[2][128 * 40];
  const int tid = threadIdx.x, lane = tid & 31, w = tid >> 5;
  const int wr = w >> 2, wc = w & 3;
  const int m0 = blockIdx.y * 128, n0 = blockIdx.x * 128;
  const int z = blockIdx.z;
  const u16* Ab = A + (long)z * sA + (long)m0 * K;
  const u16* Bb = BT + (long)z * sB + (long)n0 * K;
  const float* rp = res ? res + (long)z * sC : nullptr;
  const int nK = K >> 5;

  v8f zero = {0.f, 0.f, 0.f, 0.f, 0.f, 0.f, 0.f, 0.f};
  v8f acc[4][2];
#pragma unroll
  for (int i = 0; i < 4; ++i)
#pragma unroll
    for (int j = 0; j < 2; ++j) acc[i][j] = zero;

#if USE_TDM
  unsigned offA[2] = {(unsigned)(size_t)&lA[0][0], (unsigned)(size_t)&lA[1][0]};
  unsigned offB[2] = {(unsigned)(size_t)&lB[0][0], (unsigned)(size_t)&lB[1][0]};
  if (w == 0) {
    tdm_load_2d(offA[0], Ab, 32, 128, (unsigned long long)K);
    tdm_load_2d(offB[0], Bb, 32, 128, (unsigned long long)K);
  }
  for (int kt = 0; kt < nK; ++kt) {
    if (w == 0) {
      if (kt + 1 < nK) {
        tdm_load_2d(offA[(kt + 1) & 1], Ab + (long)(kt + 1) * 32, 32, 128, (unsigned long long)K);
        tdm_load_2d(offB[(kt + 1) & 1], Bb + (long)(kt + 1) * 32, 32, 128, (unsigned long long)K);
        __builtin_amdgcn_s_wait_tensorcnt((short)2);   // current tile complete
      } else {
        __builtin_amdgcn_s_wait_tensorcnt((short)0);
      }
    }
    __syncthreads();
    const u16* la = lA[kt & 1];
    const u16* lb = lB[kt & 1];
    v16bf fb[2];
#pragma unroll
    for (int nt = 0; nt < 2; ++nt) fb[nt] = ld_fragB(&lb[(wc * 32 + nt * 16) * 40], 40);
#pragma unroll
    for (int mt = 0; mt < 4; ++mt) {
      v16bf fa = ld_fragA(&la[(wr * 64 + mt * 16) * 40], 40);
#pragma unroll
      for (int nt = 0; nt < 2; ++nt) acc[mt][nt] = wmma_bf16(fa, fb[nt], acc[mt][nt]);
    }
    __syncthreads();
  }
#else
  for (int kt = 0; kt < nK; ++kt) {
    int c = (tid & 3) << 3, r = tid >> 2;
#pragma unroll
    for (int i = 0; i < 2; ++i) {
      *(uint4*)(&lA[0][(r + i * 64) * 40 + c]) =
          *(const uint4*)(Ab + (long)(r + i * 64) * K + kt * 32 + c);
      *(uint4*)(&lB[0][(r + i * 64) * 40 + c]) =
          *(const uint4*)(Bb + (long)(r + i * 64) * K + kt * 32 + c);
    }
    __syncthreads();
    v16bf fb[2];
#pragma unroll
    for (int nt = 0; nt < 2; ++nt) fb[nt] = ld_fragB(&lB[0][(wc * 32 + nt * 16) * 40], 40);
#pragma unroll
    for (int mt = 0; mt < 4; ++mt) {
      v16bf fa = ld_fragA(&lA[0][(wr * 64 + mt * 16) * 40], 40);
#pragma unroll
      for (int nt = 0; nt < 2; ++nt) acc[mt][nt] = wmma_bf16(fa, fb[nt], acc[mt][nt]);
    }
    __syncthreads();
  }
#endif

  // branch-hoisted epilogue with pointer marching
  const int half8 = (lane >> 4) << 3, nl = lane & 15;
  const long base = (long)(m0 + wr * 64 + half8) * N + (n0 + wc * 32 + nl);
  float* Cz = C + (long)z * sC;
  if (rp) {
#pragma unroll
    for (int mt = 0; mt < 4; ++mt)
#pragma unroll
      for (int nt = 0; nt < 2; ++nt) {
        float* cp = Cz + base + (long)mt * 16 * N + nt * 16;
        const float* rq = rp + base + (long)mt * 16 * N + nt * 16;
#pragma unroll
        for (int v = 0; v < 8; ++v) { *cp = acc[mt][nt][v] + *rq; cp += N; rq += N; }
      }
  } else {
#pragma unroll
    for (int mt = 0; mt < 4; ++mt)
#pragma unroll
      for (int nt = 0; nt < 2; ++nt) {
        float* cp = Cz + base + (long)mt * 16 * N + nt * 16;
#pragma unroll
        for (int v = 0; v < 8; ++v) { *cp = acc[mt][nt][v]; cp += N; }
      }
  }
}

// ---------------- Flash attention: per (b,h), 128 q-rows per block ----------------
// Q,K: [B,NH,N,HD] bf16 (Q pre-scaled by 1/8); VT: [B,NH,HD,N] bf16
__global__ __launch_bounds__(256) void flash_attn(
    const u16* __restrict__ Q, const u16* __restrict__ Kk, const u16* __restrict__ VT,
    float* __restrict__ O) {
  __shared__ u16 sQ[128 * 72];
  __shared__ u16 sK[32 * 72];
  __shared__ u16 sVT[64 * 40];
  __shared__ u16 sP[8 * 16 * 40];
  const int tid = threadIdx.x, lane = tid & 31, w = tid >> 5;
  const int bh = blockIdx.y, q0 = blockIdx.x * 128;
  const int b = bh >> 4, h = bh & 15;
  const long base = (long)bh * 1024 * 64;

  {
    int c = (tid & 7) << 3, r = tid >> 3;
#pragma unroll
    for (int i = 0; i < 4; ++i) {
      uint4 d = *(const uint4*)(Q + base + (long)(q0 + r + i * 32) * 64 + c);
      *(uint4*)(&sQ[(r + i * 32) * 72 + c]) = d;
    }
  }
  __syncthreads();
  v16bf qf[2];
  qf[0] = ld_fragA(&sQ[w * 16 * 72], 72);
  qf[1] = ld_fragA(&sQ[w * 16 * 72 + 32], 72);

  v8f zero = {0.f, 0.f, 0.f, 0.f, 0.f, 0.f, 0.f, 0.f};
  v8f of[4];
  float rm[8], rl[8];
#pragma unroll
  for (int nt = 0; nt < 4; ++nt) of[nt] = zero;
#pragma unroll
  for (int v = 0; v < 8; ++v) { rm[v] = -3.0e38f; rl[v] = 0.f; }

  const int half8 = (lane >> 4) << 3, nl = lane & 15;
  for (int k0 = 0; k0 < q0 + 128; k0 += 32) {
    __syncthreads();
    {
      int c = (tid & 7) << 3, r = tid >> 3;               // K tile: 32 x 64
      *(uint4*)(&sK[r * 72 + c]) = *(const uint4*)(Kk + base + (long)(k0 + r) * 64 + c);
      int c2 = (tid & 3) << 3, r2 = tid >> 2;             // VT tile: 64 x 32
      *(uint4*)(&sVT[r2 * 40 + c2]) = *(const uint4*)(VT + base + (long)r2 * 1024 + k0 + c2);
    }
    __syncthreads();

    v8f st[2];
#pragma unroll
    for (int t = 0; t < 2; ++t) {
      v16bf kf0 = ld_fragB(&sK[t * 16 * 72], 72);
      v16bf kf1 = ld_fragB(&sK[t * 16 * 72 + 32], 72);
      v8f c = zero;
      c = wmma_bf16(qf[0], kf0, c);
      c = wmma_bf16(qf[1], kf1, c);
      st[t] = c;
    }
    if (k0 + 31 > q0 + w * 16) {   // causal mask needed
#pragma unroll
      for (int t = 0; t < 2; ++t)
#pragma unroll
        for (int v = 0; v < 8; ++v) {
          int qg = q0 + w * 16 + v + half8;
          int kg = k0 + t * 16 + nl;
          if (kg > qg) st[t][v] = -3.0e38f;
        }
    }
#pragma unroll
    for (int v = 0; v < 8; ++v) {
      float mx = fmaxf(st[0][v], st[1][v]);
#pragma unroll
      for (int o = 1; o < 16; o <<= 1) mx = fmaxf(mx, __shfl_xor(mx, o));
      float mnew = fmaxf(rm[v], mx);
      float alpha = __expf(rm[v] - mnew);
      float p0 = __expf(st[0][v] - mnew);
      float p1 = __expf(st[1][v] - mnew);
      float rs = p0 + p1;
#pragma unroll
      for (int o = 1; o < 16; o <<= 1) rs += __shfl_xor(rs, o);
      rm[v] = mnew;
      rl[v] = rl[v] * alpha + rs;
#pragma unroll
      for (int nt = 0; nt < 4; ++nt) of[nt][v] *= alpha;
      sP[(w * 16 + v + half8) * 40 + nl] = f2bf(p0);
      sP[(w * 16 + v + half8) * 40 + nl + 16] = f2bf(p1);
    }
    v16bf pf = ld_fragA(&sP[w * 16 * 40], 40);
#pragma unroll
    for (int nt = 0; nt < 4; ++nt) {
      v16bf vf = ld_fragB(&sVT[nt * 16 * 40], 40);
      of[nt] = wmma_bf16(pf, vf, of[nt]);
    }
  }
#pragma unroll
  for (int nt = 0; nt < 4; ++nt) {
    float* op = O + (long)(b * 1024 + q0 + w * 16 + half8) * 1024 + h * 64 + nt * 16 + nl;
#pragma unroll
    for (int v = 0; v < 8; ++v) { *op = of[nt][v] / rl[v]; op += 1024; }
  }
}

// ---------------- RMSNorm (f32 in -> bf16 out) ----------------
__global__ __launch_bounds__(256) void rmsnorm_kernel(const float* __restrict__ X,
                                                      const float* __restrict__ Wt,
                                                      u16* __restrict__ Y) {
  __shared__ float red[8];
  int row = blockIdx.x;
  const float* x = X + (long)row * 1024;
  float s = 0.f;
  for (int i = threadIdx.x; i < 1024; i += 256) { float v = x[i]; s += v * v; }
#pragma unroll
  for (int o = 16; o > 0; o >>= 1) s += __shfl_xor(s, o);
  int lane = threadIdx.x & 31, w = threadIdx.x >> 5;
  if (lane == 0) red[w] = s;
  __syncthreads();
  if (w == 0) {
    float t = (lane < 8) ? red[lane] : 0.f;
#pragma unroll
    for (int o = 4; o > 0; o >>= 1) t += __shfl_xor(t, o);
    if (lane == 0) red[0] = t;
  }
  __syncthreads();
  float r = rsqrtf(red[0] * (1.0f / 1024.0f) + 1e-5f);
  u16* y = Y + (long)row * 1024;
  for (int i = threadIdx.x; i < 1024; i += 256) y[i] = f2bf(x[i] * r * Wt[i]);
}

// ---------------- RoPE + repack: Q,K -> [B,NH,N,HD]; V -> [B,NH,HD,N] (transposed) ----------------
__global__ void rope_pack(const float* __restrict__ qkv, const float* __restrict__ Cb,
                          const float* __restrict__ Sb, u16* __restrict__ Qb,
                          u16* __restrict__ Kb, u16* __restrict__ Vb) {
  long idx = (long)blockIdx.x * blockDim.x + threadIdx.x;
  if (idx >= (long)4 * 1024 * 16 * 32) return;
  int p = (int)(idx & 31);
  int h = (int)((idx >> 5) & 15);
  int n = (int)((idx >> 9) & 1023);
  int b = (int)(idx >> 19);
  long row = (long)(b * 1024 + n) * 3072;
  float c = Cb[n * 32 + p], s = Sb[n * 32 + p];
  float q0 = qkv[row + h * 64 + 2 * p], q1 = qkv[row + h * 64 + 2 * p + 1];
  float k0 = qkv[row + 1024 + h * 64 + 2 * p], k1 = qkv[row + 1024 + h * 64 + 2 * p + 1];
  float v0 = qkv[row + 2048 + h * 64 + 2 * p], v1 = qkv[row + 2048 + h * 64 + 2 * p + 1];
  long o = ((long)(b * 16 + h) * 1024 + n) * 64 + 2 * p;
  Qb[o] = f2bf((q0 * c - q1 * s) * 0.125f);
  Qb[o + 1] = f2bf((q0 * s + q1 * c) * 0.125f);
  Kb[o] = f2bf(k0 * c - k1 * s);
  Kb[o + 1] = f2bf(k0 * s + k1 * c);
  long ov = ((long)(b * 16 + h) * 64 + 2 * p) * 1024 + n;   // transposed V
  Vb[ov] = f2bf(v0);
  Vb[ov + 1024] = f2bf(v1);
}

// ---------------- elementwise packs ----------------
__global__ void pack_bf(const float* __restrict__ in, u16* __restrict__ out, long n) {
  long i = (long)blockIdx.x * blockDim.x + threadIdx.x;
  if (i < n) out[i] = f2bf(in[i]);
}
// transpose-pack: in [z][Kd][Nd] f32 -> out [z][Nd][Kd] bf16
__global__ __launch_bounds__(256) void pack_bf_transpose(const float* __restrict__ in,
                                                         u16* __restrict__ out,
                                                         int Kd, int Nd) {
  __shared__ float t[32][33];
  int z = blockIdx.z;
  int k0 = blockIdx.y * 32, n0 = blockIdx.x * 32;
  const float* ip = in + (long)z * Kd * Nd;
  u16* op = out + (long)z * Kd * Nd;
  int lx = threadIdx.x & 31, ly = threadIdx.x >> 5;
#pragma unroll
  for (int i = 0; i < 4; ++i)
    t[ly + i * 8][lx] = ip[(long)(k0 + ly + i * 8) * Nd + n0 + lx];
  __syncthreads();
#pragma unroll
  for (int i = 0; i < 4; ++i)
    op[(long)(n0 + ly + i * 8) * Kd + k0 + lx] = f2bf(t[lx][ly + i * 8]);
}
__global__ void silu_mul_pack(const float* __restrict__ h1, const float* __restrict__ h3,
                              u16* __restrict__ out, long n) {
  long i = (long)blockIdx.x * blockDim.x + threadIdx.x;
  if (i < n) { float a = h1[i]; out[i] = f2bf((a / (1.f + __expf(-a))) * h3[i]); }
}
__global__ void gelu_pack(const float* __restrict__ h, u16* __restrict__ out, long n) {
  long i = (long)blockIdx.x * blockDim.x + threadIdx.x;
  if (i < n) { float a = h[i]; out[i] = f2bf(0.5f * a * (1.f + erff(a * 0.70710678118f))); }
}
__global__ void copy_f32(const float* __restrict__ in, float* __restrict__ out, long n) {
  long i = (long)blockIdx.x * blockDim.x + threadIdx.x;
  if (i < n) out[i] = in[i];
}
__global__ void zero_f32(float* p, int n) {
  int i = blockIdx.x * blockDim.x + threadIdx.x;
  if (i < n) p[i] = 0.f;
}

// ---------------- MoE ----------------
__device__ inline float urand(unsigned li, unsigned tok) {
  unsigned h = tok * 0x9E3779B9u + li * 0x85EBCA6Bu + 0x27D4EB2Fu;
  h ^= h >> 16; h *= 0x7FEB352Du; h ^= h >> 15; h *= 0x846CA68Bu; h ^= h >> 16;
  return (float)(h >> 8) * (1.0f / 16777216.0f);
}

__global__ __launch_bounds__(256) void router_gemm(const u16* __restrict__ Xn,
                                                   const float* __restrict__ Wg,
                                                   float* __restrict__ L) {
  int tok = blockIdx.x * 8 + (threadIdx.x >> 5);
  int lane = threadIdx.x & 31;
  float acc[8] = {0, 0, 0, 0, 0, 0, 0, 0};
  const u16* x = Xn + (long)tok * 1024;
  for (int d = lane; d < 1024; d += 32) {
    float xv = bf2f(x[d]);
#pragma unroll
    for (int e = 0; e < 8; ++e) acc[e] += xv * Wg[d * 8 + e];
  }
#pragma unroll
  for (int e = 0; e < 8; ++e)
#pragma unroll
    for (int o = 16; o > 0; o >>= 1) acc[e] += __shfl_xor(acc[e], o);
  if (lane == 0)
    for (int e = 0; e < 8; ++e) L[tok * 8 + e] = acc[e];
}

__global__ void route_kernel(const float* __restrict__ L, int* idx1, int* idx2,
                             float* g1o, float* g2o, int* keep2, int* pos1, int* pos2,
                             float* rawsum, float* cnt1, int li) {
  int tok = blockIdx.x * blockDim.x + threadIdx.x;
  if (tok >= 4096) return;
  float lg[8];
  float mx = -3e38f;
#pragma unroll
  for (int e = 0; e < 8; ++e) { lg[e] = L[tok * 8 + e]; mx = fmaxf(mx, lg[e]); }
  float sum = 0.f;
#pragma unroll
  for (int e = 0; e < 8; ++e) { lg[e] = __expf(lg[e] - mx); sum += lg[e]; }
  float inv = 1.f / sum;
#pragma unroll
  for (int e = 0; e < 8; ++e) lg[e] *= inv;
  int i1 = 0; float g1 = lg[0];
#pragma unroll
  for (int e = 1; e < 8; ++e) if (lg[e] > g1) { g1 = lg[e]; i1 = e; }
  int i2 = (i1 == 0) ? 1 : 0; float g2 = lg[i2];
#pragma unroll
  for (int e = 0; e < 8; ++e) if (e != i1 && lg[e] > g2) { g2 = lg[e]; i2 = e; }
  float denom = g1 + g2 + 1e-9f;
  float g1n = g1 / denom, g2n = g2 / denom;
  int k2 = (urand((unsigned)li, (unsigned)tok) < g2n / 0.2f) ? 1 : 0;
  idx1[tok] = i1; idx2[tok] = i2;
  g1o[tok] = g1n; g2o[tok] = g2n; keep2[tok] = k2;
  pos1[tok] = -1; pos2[tok] = -1;
  int b = tok >> 10;
#pragma unroll
  for (int e = 0; e < 8; ++e) atomicAdd(&rawsum[b * 8 + e], lg[e]);
  atomicAdd(&cnt1[b * 8 + i1], 1.0f);
}

// one wave per (b, expert): wave32 ballot prefix-sum over token order
__global__ __launch_bounds__(32) void scan_kernel(const int* __restrict__ idx1,
                                                  const int* __restrict__ idx2,
                                                  const int* __restrict__ keep2,
                                                  int* pos1, int* pos2, int* slot_tok) {
  int e = blockIdx.x & 7, b = blockIdx.x >> 3;
  int lane = threadIdx.x;
  for (int c = lane; c < 160; c += 32) slot_tok[(e * 4 + b) * 160 + c] = -1;
  int basec = 0;
  for (int t0 = 0; t0 < 1024; t0 += 32) {
    int tg = b * 1024 + t0 + lane;
    bool c1 = (idx1[tg] == e);
    unsigned mask = (unsigned)__ballot(c1);
    int pre = __popc(mask & ((1u << lane) - 1u));
    if (c1) {
      int p = basec + pre;
      if (p < 160) { pos1[tg] = p; slot_tok[(e * 4 + b) * 160 + p] = tg; }
    }
    basec += __popc(mask);
  }
  int base2 = basec < 160 ? basec : 160;
  for (int t0 = 0; t0 < 1024; t0 += 32) {
    int tg = b * 1024 + t0 + lane;
    bool c2 = (idx2[tg] == e) && (keep2[tg] != 0);
    unsigned mask = (unsigned)__ballot(c2);
    int pre = __popc(mask & ((1u << lane) - 1u));
    if (c2) {
      int p = base2 + pre;
      if (p < 160) { pos2[tg] = p; slot_tok[(e * 4 + b) * 160 + p] = tg; }
    }
    base2 += __popc(mask);
  }
}

__global__ __launch_bounds__(256) void gather_kernel(const int* __restrict__ slot_tok,
                                                     const u16* __restrict__ xn,
                                                     u16* __restrict__ ein) {
  int slot = blockIdx.x;
  int tok = slot_tok[slot];
  u16* dst = ein + (long)slot * 1024;
  if (tok < 0) {
    for (int i = threadIdx.x; i < 1024; i += 256) dst[i] = 0;
  } else {
    const u16* src = xn + (long)tok * 1024;
    for (int i = threadIdx.x; i < 1024; i += 256) dst[i] = src[i];
  }
}

__global__ __launch_bounds__(256) void scatter_kernel(const float* __restrict__ eout,
                                                      const int* idx1, const int* idx2,
                                                      const int* pos1, const int* pos2,
                                                      const float* g1, const float* g2,
                                                      float* __restrict__ xb) {
  int tok = blockIdx.x;
  int b = tok >> 10;
  int e1 = idx1[tok], p1 = pos1[tok];
  int e2 = idx2[tok], p2 = pos2[tok];
  float w1 = g1[tok], w2 = g2[tok];
  const float* r1 = (p1 >= 0) ? eout + ((long)(e1 * 4 + b) * 160 + p1) * 1024 : nullptr;
  const float* r2 = (p2 >= 0) ? eout + ((long)(e2 * 4 + b) * 160 + p2) * 1024 : nullptr;
  for (int i = threadIdx.x; i < 1024; i += 256) {
    float a = xb[(long)tok * 1024 + i];
    if (r1) a += w1 * r1[i];
    if (r2) a += w2 * r2[i];
    xb[(long)tok * 1024 + i] = a;
  }
}

__global__ __launch_bounds__(32) void loss_kernel(const float* rawsum, const float* cnt1,
                                                  float* lossv, int li) {
  int lane = threadIdx.x;
  float v = rawsum[lane] * cnt1[lane];
#pragma unroll
  for (int o = 16; o > 0; o >>= 1) v += __shfl_xor(v, o);
  if (lane == 0)
    lossv[li] = (v / 32.0f) * (1.0f / (1024.f * 1024.f)) * 64.0f * 0.01f;
}

__global__ void final_kernel(const float* __restrict__ xb, const float* __restrict__ lossv,
                             float* __restrict__ out) {
  long i = (long)blockIdx.x * blockDim.x + threadIdx.x;
  if (i < 4194304L) out[i] = xb[i];
  else if (i == 4194304L) out[i] = lossv[0] + lossv[1];
}

// ---------------- host orchestration ----------------
extern "C" void kernel_launch(void* const* d_in, const int* in_sizes, int n_in,
                              void* d_out, int out_size, void* d_ws, size_t ws_size,
                              hipStream_t stream) {
  (void)in_sizes; (void)n_in; (void)out_size; (void)ws_size;
  const float* x_in = (const float*)d_in[0];
  const float* fcos = (const float*)d_in[1];
  const float* fsin = (const float*)d_in[2];
  const float* lnw  = (const float*)d_in[3];
  const float* wqkv = (const float*)d_in[4];
  const float* wo   = (const float*)d_in[5];
  const float* w1   = (const float*)d_in[6];
  const float* w2   = (const float*)d_in[7];
  const float* w3   = (const float*)d_in[8];
  const float* wg   = (const float*)d_in[9];
  const float* mw1  = (const float*)d_in[10];
  const float* mw2  = (const float*)d_in[11];

  char* p = (char*)d_ws;
  auto alloc = [&](size_t bytes) -> char* {
    char* r = p; p += (bytes + 255) & ~(size_t)255; return r;
  };
  float* xbuf   = (float*)alloc(4096L * 1024 * 4);
  u16*   xn     = (u16*)alloc(4096L * 1024 * 2);
  u16*   wqkv_b = (u16*)alloc(3L * 1024 * 3072 * 2);   // transposed [3][3072][1024]
  u16*   wo_b   = (u16*)alloc(3L * 1024 * 1024 * 2);   // transposed
  u16*   w1_b   = (u16*)alloc(3L * 1024 * 2816 * 2);   // transposed [3][2816][1024]
  u16*   w3_b   = (u16*)alloc(3L * 1024 * 2816 * 2);   // transposed
  u16*   w2_b   = (u16*)alloc(3L * 2816 * 1024 * 2);   // transposed [3][1024][2816]
  u16*   mw1_b  = (u16*)alloc(2L * 8 * 1024 * 2048 * 2); // transposed [2][8][2048][1024]
  u16*   mw2_b  = (u16*)alloc(2L * 8 * 2048 * 1024 * 2); // transposed [2][8][1024][2048]
  float* big0   = (float*)alloc(4096L * 3072 * 4);     // qkv | h3 | moe hid
  float* big1   = (float*)alloc(4096L * 2816 * 4);     // h1 | (eout + hid_bf)
  u16*   qb     = (u16*)alloc(4096L * 1024 * 2);
  u16*   kb     = (u16*)alloc(4096L * 1024 * 2);
  u16*   vtb    = (u16*)alloc(4096L * 1024 * 2);       // V transposed per (b,h)
  float* attno  = (float*)alloc(4096L * 1024 * 4);
  u16*   tmpb   = (u16*)alloc(4096L * 2816 * 2);
  u16*   einb   = (u16*)alloc(8L * 640 * 1024 * 2);
  float* logits = (float*)alloc(4096L * 8 * 4);
  int*   idx1   = (int*)alloc(4096 * 4);
  int*   idx2   = (int*)alloc(4096 * 4);
  int*   pos1   = (int*)alloc(4096 * 4);
  int*   pos2   = (int*)alloc(4096 * 4);
  int*   keep2  = (int*)alloc(4096 * 4);
  float* gate1  = (float*)alloc(4096 * 4);
  float* gate2  = (float*)alloc(4096 * 4);
  int*   slot   = (int*)alloc(5120 * 4);
  float* rawsum = (float*)alloc(32 * 4);
  float* cnt1v  = (float*)alloc(32 * 4);
  float* lossv  = (float*)alloc(2 * 4);
  float* eoutp  = big1;                                // 8*640*1024 f32
  u16*   hidb   = (u16*)((char*)big1 + 8L * 640 * 1024 * 4);

  auto packN = [&](const float* in, u16* out, long n) {
    pack_bf<<<dim3((unsigned)((n + 255) / 256)), dim3(256), 0, stream>>>(in, out, n);
  };
  auto packT = [&](const float* in, u16* out, int Kd, int Nd, int batch) {
    pack_bf_transpose<<<dim3(Nd / 32, Kd / 32, batch), dim3(256), 0, stream>>>(in, out, Kd, Nd);
  };
  auto gemm = [&](const u16* A, const u16* Bw, float* C, int M, int Nc, int K,
                  long sA, long sB, long sC, int batch, const float* res) {
    gemm_bf16_wmma<<<dim3(Nc / 128, M / 128, batch), dim3(256), 0, stream>>>(
        A, Bw, C, M, Nc, K, sA, sB, sC, res);
  };

  // ---- convert all weights to bf16, transposed to [N][K] ----
  packT(wqkv, wqkv_b, 1024, 3072, 3);
  packT(wo,   wo_b,   1024, 1024, 3);
  packT(w1,   w1_b,   1024, 2816, 3);
  packT(w3,   w3_b,   1024, 2816, 3);
  packT(w2,   w2_b,   2816, 1024, 3);
  packT(mw1,  mw1_b,  1024, 2048, 16);
  packT(mw2,  mw2_b,  2048, 1024, 16);
  copy_f32<<<dim3((4194304 + 255) / 256), dim3(256), 0, stream>>>(x_in, xbuf, 4194304L);

  auto do_attn = [&](int lnIdx, int wIdx) {
    rmsnorm_kernel<<<4096, 256, 0, stream>>>(xbuf, lnw + lnIdx * 1024, xn);
    gemm(xn, wqkv_b + (long)wIdx * 1024 * 3072, big0, 4096, 3072, 1024, 0, 0, 0, 1, nullptr);
    rope_pack<<<dim3((4 * 1024 * 16 * 32) / 256), dim3(256), 0, stream>>>(big0, fcos, fsin, qb, kb, vtb);
    flash_attn<<<dim3(8, 64), dim3(256), 0, stream>>>(qb, kb, vtb, attno);
    packN(attno, tmpb, 4096L * 1024);
    gemm(tmpb, wo_b + (long)wIdx * 1024 * 1024, xbuf, 4096, 1024, 1024, 0, 0, 0, 1, xbuf);
  };
  auto do_mlp = [&](int lnIdx, int wIdx) {
    rmsnorm_kernel<<<4096, 256, 0, stream>>>(xbuf, lnw + lnIdx * 1024, xn);
    gemm(xn, w1_b + (long)wIdx * 1024 * 2816, big1, 4096, 2816, 1024, 0, 0, 0, 1, nullptr);
    gemm(xn, w3_b + (long)wIdx * 1024 * 2816, big0, 4096, 2816, 1024, 0, 0, 0, 1, nullptr);
    long n = 4096L * 2816;
    silu_mul_pack<<<dim3((unsigned)((n + 255) / 256)), dim3(256), 0, stream>>>(big1, big0, tmpb, n);
    gemm(tmpb, w2_b + (long)wIdx * 2816 * 1024, xbuf, 4096, 1024, 2816, 0, 0, 0, 1, xbuf);
  };
  auto do_moe = [&](int lnIdx, int li) {
    rmsnorm_kernel<<<4096, 256, 0, stream>>>(xbuf, lnw + lnIdx * 1024, xn);
    zero_f32<<<1, 64, 0, stream>>>(rawsum, 32);
    zero_f32<<<1, 64, 0, stream>>>(cnt1v, 32);
    router_gemm<<<512, 256, 0, stream>>>(xn, wg + (long)li * 8192, logits);
    route_kernel<<<16, 256, 0, stream>>>(logits, idx1, idx2, gate1, gate2, keep2,
                                         pos1, pos2, rawsum, cnt1v, li);
    scan_kernel<<<32, 32, 0, stream>>>(idx1, idx2, keep2, pos1, pos2, slot);
    gather_kernel<<<5120, 256, 0, stream>>>(slot, xn, einb);
    gemm(einb, mw1_b + (long)li * 8 * 1024 * 2048, big0, 640, 2048, 1024,
         640L * 1024, 1024L * 2048, 640L * 2048, 8, nullptr);
    long n = 8L * 640 * 2048;
    gelu_pack<<<dim3((unsigned)((n + 255) / 256)), dim3(256), 0, stream>>>(big0, hidb, n);
    gemm(hidb, mw2_b + (long)li * 8 * 2048 * 1024, eoutp, 640, 1024, 2048,
         640L * 2048, 2048L * 1024, 640L * 1024, 8, nullptr);
    scatter_kernel<<<4096, 256, 0, stream>>>(eoutp, idx1, idx2, pos1, pos2, gate1, gate2, xbuf);
    loss_kernel<<<1, 32, 0, stream>>>(rawsum, cnt1v, lossv, li);
  };

  do_attn(0, 0);
  do_mlp(1, 0);
  do_moe(2, 0);
  do_mlp(3, 1);
  do_attn(4, 1);
  do_moe(5, 1);
  do_attn(6, 2);
  do_mlp(7, 2);
  final_kernel<<<dim3((4194305 + 255) / 256), dim3(256), 0, stream>>>(xbuf, lossv, (float*)d_out);
}
